// FlashAttention_27685359190300
// MI455X (gfx1250) — matrix operand order repaired and checked by reading, not compiled, not run on hardware
//
#include <hip/hip_runtime.h>
#include <math.h>

typedef __attribute__((ext_vector_type(16))) __bf16 v16bf;
typedef __attribute__((ext_vector_type(8)))  float  v8f;

#define N_CTX   8192
#define DHEAD   128
#define BN      32            // keys per iteration
#define NWAVES  8
#define ROWS_W  32            // query rows per wave (2 m-tiles of 16)
#define BM      (NWAVES * ROWS_W)   // 256 query rows per block

union FragBF { v16bf v; unsigned u[8]; };

__device__ __forceinline__ unsigned short f2bf(float x) {
    __bf16 b = (__bf16)x;
    return __builtin_bit_cast(unsigned short, b);
}
__device__ __forceinline__ unsigned pk2(float a, float b) {
#if __has_builtin(__builtin_amdgcn_cvt_pk_bf16_f32)
    auto r = __builtin_amdgcn_cvt_pk_bf16_f32(a, b);
    if constexpr (sizeof(r) == 4) {
        unsigned u; __builtin_memcpy(&u, &r, 4); return u;
    } else {
        return (unsigned)f2bf(a) | ((unsigned)f2bf(b) << 16);
    }
#else
    return (unsigned)f2bf(a) | ((unsigned)f2bf(b) << 16);
#endif
}

__global__ __launch_bounds__(256)
void fa_fwd_bf16_wmma_t(const float* __restrict__ Qg, const float* __restrict__ Kg,
                        const float* __restrict__ Vg, float* __restrict__ Og) {
    const int tid  = threadIdx.x;
    const int lane = tid & 31;
    const int wave = tid >> 5;
    const int lh   = lane >> 4;     // lane half
    const int idx  = lane & 15;

    __shared__ __align__(16) unsigned short Klds[BN * DHEAD];         // bf16 [key][dim]
    __shared__ __align__(16) unsigned short Vtlds[DHEAD * BN];        // bf16 [dim][key]
    __shared__ __align__(16) unsigned short Plds[NWAVES][2][16 * BN]; // bf16 P [query][key]

    // K-offset patterns for 16-bit fragments (ISA 7.12.2):
    // A (16x32): interleaved  half0: {0,2,4,6,16,18,20,22}  half1: +8
    // B (32x16): interleaved  half0: {0,2,4,6,16,18,20,22}  half1: +8
    int kpA[8], kpB[8];
#pragma unroll
    for (int p = 0; p < 8; ++p) {
        kpA[p] = 2 * p + ((p >= 4) ? 8 : 0) + 8 * lh;
        kpB[p] = 2 * p + ((p >= 4) ? 8 : 0) + 8 * lh;
    }

    const int qrow0 = blockIdx.x * BM + wave * ROWS_W;
    const float LOG2E = 1.4426950408889634f;

    // Cooperative-load assignment: each thread owns 2 adjacent rows x 8 cols.
    const int cr0 = (tid >> 4) * 2;       // rows {0,2,...,30}
    const int cc0 = (tid & 15) * 8;       // cols {0,8,...,120}

    // ---- Q^T as B-fragments (pre-scaled by log2e), loaded once ----
    v16bf bq[2][4];
#pragma unroll
    for (int mt = 0; mt < 2; ++mt) {
        const float* qrow = Qg + (size_t)(qrow0 + mt * 16 + idx) * DHEAD;
#pragma unroll
        for (int kt = 0; kt < 4; ++kt)
#pragma unroll
            for (int p = 0; p < 8; ++p) {
                int k = kt * 32 + kpB[p];
                bq[mt][kt][2 * p]     = (__bf16)(qrow[k]     * LOG2E);
                bq[mt][kt][2 * p + 1] = (__bf16)(qrow[k + 1] * LOG2E);
            }
    }

    // ---- O^T accumulators + per-lane softmax state (lane == query column) ----
    v8f acc[2][8];
    float m_[2], l_[2];
#pragma unroll
    for (int mt = 0; mt < 2; ++mt) {
#pragma unroll
        for (int nt = 0; nt < 8; ++nt) acc[mt][nt] = (v8f){0,0,0,0,0,0,0,0};
        m_[mt] = -INFINITY; l_[mt] = 0.0f;
    }

    for (int kb = 0; kb < N_CTX; kb += BN) {
        __syncthreads();
        // ---- Cooperative staging: 8x global_load_b128, packed bf16 stores ----
        {
            const float4* k4  = (const float4*)&Kg[(size_t)(kb + cr0) * DHEAD + cc0];
            const float4* v4  = (const float4*)&Vg[(size_t)(kb + cr0) * DHEAD + cc0];
            const float4* k4b = (const float4*)&Kg[(size_t)(kb + cr0 + 1) * DHEAD + cc0];
            const float4* v4b = (const float4*)&Vg[(size_t)(kb + cr0 + 1) * DHEAD + cc0];
            float4 ka0 = k4[0],  ka1 = k4[1];
            float4 kb0 = k4b[0], kb1 = k4b[1];
            float4 va0 = v4[0],  va1 = v4[1];
            float4 vb0 = v4b[0], vb1 = v4b[1];
            uint4 kr0 = { pk2(ka0.x, ka0.y), pk2(ka0.z, ka0.w),
                          pk2(ka1.x, ka1.y), pk2(ka1.z, ka1.w) };
            uint4 kr1 = { pk2(kb0.x, kb0.y), pk2(kb0.z, kb0.w),
                          pk2(kb1.x, kb1.y), pk2(kb1.z, kb1.w) };
            *(uint4*)&Klds[(cr0)     * DHEAD + cc0] = kr0;
            *(uint4*)&Klds[(cr0 + 1) * DHEAD + cc0] = kr1;
            float va[8] = { va0.x, va0.y, va0.z, va0.w, va1.x, va1.y, va1.z, va1.w };
            float vb[8] = { vb0.x, vb0.y, vb0.z, vb0.w, vb1.x, vb1.y, vb1.z, vb1.w };
#pragma unroll
            for (int j = 0; j < 8; ++j)
                *(unsigned*)&Vtlds[(cc0 + j) * BN + cr0] = pk2(va[j], vb[j]);
        }
        if (kb + BN < N_CTX) {  // global_prefetch_b8 for next block
            __builtin_prefetch(&Kg[(size_t)(kb + BN + cr0) * DHEAD + cc0], 0, 0);
            __builtin_prefetch(&Vg[(size_t)(kb + BN + cr0) * DHEAD + cc0], 0, 0);
        }
        __syncthreads();

        // ---- S^T = K_blk * Q^T  (two 16x16 C tiles per m-tile: kh = key half) ----
        v8f ST[2][2];   // [mt][kh]; lane holds query=idx, keys kh*16 + 8*lh + r
#pragma unroll
        for (int mt = 0; mt < 2; ++mt) { ST[mt][0] = (v8f){0,0,0,0,0,0,0,0};
                                         ST[mt][1] = (v8f){0,0,0,0,0,0,0,0}; }
#pragma unroll
        for (int kt = 0; kt < 4; ++kt) {
            FragBF ak0, ak1;   // A-fragments: K rows (keys) x dims
#pragma unroll
            for (int p = 0; p < 8; ++p) {
                int dd = kt * 32 + kpA[p];
                ak0.u[p] = *(const unsigned*)&Klds[(idx)      * DHEAD + dd];
                ak1.u[p] = *(const unsigned*)&Klds[(16 + idx) * DHEAD + dd];
            }
#pragma unroll
            for (int mt = 0; mt < 2; ++mt) {
                ST[mt][0] = __builtin_amdgcn_wmma_f32_16x16x32_bf16(
                                false, ak0.v, false, bq[mt][kt], (short)0, ST[mt][0], false, false);
                ST[mt][1] = __builtin_amdgcn_wmma_f32_16x16x32_bf16(
                                false, ak1.v, false, bq[mt][kt], (short)0, ST[mt][1], false, false);
            }
        }

        // ---- Online softmax: per-lane scalar stats, 1 cross-half shuffle each ----
#pragma unroll
        for (int mt = 0; mt < 2; ++mt) {
            float vmax = ST[mt][0][0];
#pragma unroll
            for (int r = 1; r < 8; ++r) vmax = fmaxf(vmax, ST[mt][0][r]);
#pragma unroll
            for (int r = 0; r < 8; ++r) vmax = fmaxf(vmax, ST[mt][1][r]);
            vmax = fmaxf(vmax, __shfl_xor(vmax, 16));
            float mold = m_[mt];
            float mnew = fmaxf(mold, vmax);
            // Rescale only when some lane's running max actually moved.
            if (__any(mnew > mold)) {
                float scale = __builtin_amdgcn_exp2f(mold - mnew);
                l_[mt] *= scale;
#pragma unroll
                for (int nt = 0; nt < 8; ++nt)
#pragma unroll
                    for (int r = 0; r < 8; ++r) acc[mt][nt][r] *= scale;
            }
            m_[mt] = mnew;
#pragma unroll
            for (int kh = 0; kh < 2; ++kh)
#pragma unroll
                for (int r = 0; r < 8; ++r)
                    ST[mt][kh][r] = __builtin_amdgcn_exp2f(ST[mt][kh][r] - mnew);
            // Tree-structured sum: 4-deep critical path instead of 16.
            float a0 = ST[mt][0][0] + ST[mt][0][1], a1 = ST[mt][0][2] + ST[mt][0][3];
            float a2 = ST[mt][0][4] + ST[mt][0][5], a3 = ST[mt][0][6] + ST[mt][0][7];
            float a4 = ST[mt][1][0] + ST[mt][1][1], a5 = ST[mt][1][2] + ST[mt][1][3];
            float a6 = ST[mt][1][4] + ST[mt][1][5], a7 = ST[mt][1][6] + ST[mt][1][7];
            float b0 = a0 + a1, b1 = a2 + a3, b2 = a4 + a5, b3 = a6 + a7;
            float rs = (b0 + b1) + (b2 + b3);
            rs += __shfl_xor(rs, 16);
            l_[mt] += rs;

            // Store P as [query][key] bf16: 8 consecutive keys per kh -> one b128 store
#pragma unroll
            for (int kh = 0; kh < 2; ++kh) {
                uint4 pkv = { pk2(ST[mt][kh][0], ST[mt][kh][1]),
                              pk2(ST[mt][kh][2], ST[mt][kh][3]),
                              pk2(ST[mt][kh][4], ST[mt][kh][5]),
                              pk2(ST[mt][kh][6], ST[mt][kh][7]) };
                *(uint4*)&Plds[wave][mt][idx * BN + kh * 16 + 8 * lh] = pkv;
            }
        }

        // ---- P^T as B-fragments (same-wave DS ordering; no barrier needed) ----
        FragBF ptb[2];
#pragma unroll
        for (int mt = 0; mt < 2; ++mt)
#pragma unroll
            for (int p = 0; p < 8; ++p)
                ptb[mt].u[p] = *(const unsigned*)&Plds[wave][mt][idx * BN + kpB[p]];

        // ---- O^T += V^T * P^T ----
#pragma unroll
        for (int nt = 0; nt < 8; ++nt) {
            FragBF av;   // A-fragment: V^T tile (dims x keys)
#pragma unroll
            for (int p = 0; p < 8; ++p)
                av.u[p] = *(const unsigned*)&Vtlds[(nt * 16 + idx) * BN + kpA[p]];
#pragma unroll
            for (int mt = 0; mt < 2; ++mt)
                acc[mt][nt] = __builtin_amdgcn_wmma_f32_16x16x32_bf16(
                                  false, av.v, false, ptb[mt].v, (short)0, acc[mt][nt], false, false);
        }
    }

    // ---- Epilogue: O = O^T / l, 8 contiguous dims per lane -> b128 stores ----
#pragma unroll
    for (int mt = 0; mt < 2; ++mt) {
        float inv = 1.0f / l_[mt];
        size_t row = (size_t)(qrow0 + mt * 16 + idx);
#pragma unroll
        for (int nt = 0; nt < 8; ++nt) {
            float* dst = &Og[row * DHEAD + nt * 16 + 8 * lh];
            float4 lo = { acc[mt][nt][0] * inv, acc[mt][nt][1] * inv,
                          acc[mt][nt][2] * inv, acc[mt][nt][3] * inv };
            float4 hi = { acc[mt][nt][4] * inv, acc[mt][nt][5] * inv,
                          acc[mt][nt][6] * inv, acc[mt][nt][7] * inv };
            *(float4*)dst       = lo;
            *(float4*)(dst + 4) = hi;
        }
    }
}

extern "C" void kernel_launch(void* const* d_in, const int* in_sizes, int n_in,
                              void* d_out, int out_size, void* d_ws, size_t ws_size,
                              hipStream_t stream) {
    (void)in_sizes; (void)n_in; (void)out_size; (void)d_ws; (void)ws_size;
    const float* Q = (const float*)d_in[0];
    const float* K = (const float*)d_in[1];
    const float* V = (const float*)d_in[2];
    float* O = (float*)d_out;
    dim3 grid(N_CTX / BM), block(256);
    fa_fwd_bf16_wmma_t<<<grid, block, 0, stream>>>(Q, K, V, O);
}
